// GridRenderer_12421045420387
// MI455X (gfx1250) — matrix operand order repaired and checked by reading, not compiled, not run on hardware
//
#include <hip/hip_runtime.h>

typedef __attribute__((ext_vector_type(16))) _Float16 v16h;
typedef __attribute__((ext_vector_type(8)))  _Float16 v8h;
typedef __attribute__((ext_vector_type(8)))  float    v8f;

#define TBL_SIZE  (1u << 19)
#define TBL_MASK  (TBL_SIZE - 1u)
#define PRIME1    2654435761u
#define PRIME2    805459861u

#if __has_builtin(__builtin_amdgcn_fmed3f)
__device__ __forceinline__ float relu_f(float v) {
    return __builtin_amdgcn_fmed3f(v, 0.0f, 3.0e38f);   // single v_med3_f32
}
#else
__device__ __forceinline__ float relu_f(float v) { return fmaxf(v, 0.0f); }
#endif

// per-lane LDS stride for weight fragments: 200 halves = 100 dwords (odd*4)
// -> b128 reloads hit each bank exactly twice (optimal for 128 dwords/wave)
#define WSTRIDE 200

__global__ __launch_bounds__(256) void
gridnet_sigma_kernel(const float* __restrict__ x,
                     const float* __restrict__ cc,
                     const float* __restrict__ tables,
                     const float* __restrict__ W0,
                     const float* __restrict__ W1,
                     const float* __restrict__ W2,
                     float* __restrict__ out,
                     int n, int numTiles)
{
    // resolutions: floor(16 * b^l), b = (2048/16)^(1/15)
    static constexpr int kRes[16] = {16, 22, 30, 42, 58, 80, 111, 153,
                                     212, 294, 406, 561, 776, 1072, 1482, 2048};

    const int lane    = threadIdx.x & 31;
    const int nIdx    = lane & 15;         // point-row within tile / N-column within tile
    const bool loHalf = lane < 16;
    const int waveId  = threadIdx.x >> 5;
    const int rowOff  = loHalf ? 0 : 8;    // C/D row offset for this lane half
    const int kb0     = loHalf ? 0 : 16;   // B-fragment K base for this lane half
    const int cbase   = loHalf ? 0 : 16;   // A-fragment K base (layer 1 reload)

    // ---- LDS: weight fragments (per-lane, shared by all waves) + h0 staging ----
    __shared__ __align__(16) _Float16 ldsW[32 * WSTRIDE];            // ~12.5 KB
    __shared__ __align__(16) _Float16 ldsH[8][64 * 16];              // 16 KB, col-major [col][row]

    // ---- wave 0 builds the 12 weight B-fragments once, stores to LDS ----
    // B 32x16 fragments use contiguous k (A matches): lanes 0-15 hold K=0..15 (2/VGPR), lanes 16-31 hold K=16..31.
    if (waveId == 0) {
#pragma unroll
        for (int t = 0; t < 4; ++t) {
            const int nc = t * 16 + nIdx;
            v16h f0, flo, fhi;
#pragma unroll
            for (int v = 0; v < 8; ++v) {
                f0[2 * v]      = (_Float16)W0[(kb0 + 2 * v) * 64 + nc];
                f0[2 * v + 1]  = (_Float16)W0[(kb0 + 2 * v + 1) * 64 + nc];
                flo[2 * v]     = (_Float16)W1[(kb0 + 2 * v) * 64 + nc];
                flo[2 * v + 1] = (_Float16)W1[(kb0 + 2 * v + 1) * 64 + nc];
                fhi[2 * v]     = (_Float16)W1[(32 + kb0 + 2 * v) * 64 + nc];
                fhi[2 * v + 1] = (_Float16)W1[(32 + kb0 + 2 * v + 1) * 64 + nc];
            }
            *(v16h*)&ldsW[lane * WSTRIDE + t * 16]       = f0;
            *(v16h*)&ldsW[lane * WSTRIDE + (4 + t) * 16] = flo;
            *(v16h*)&ldsW[lane * WSTRIDE + (8 + t) * 16] = fhi;
        }
    }
    __syncthreads();

    float w2c[4];
#pragma unroll
    for (int t = 0; t < 4; ++t) w2c[t] = W2[(t * 16 + nIdx) * 65];   // column 0 of W2

    const float c0 = cc[0], c1 = cc[1], c2 = cc[2];
    const float2* __restrict__ tbl2 = (const float2*)tables;
    const _Float16* wl = &ldsW[lane * WSTRIDE];
    _Float16* st = &ldsH[waveId][0];

    const int totalWaves = gridDim.x * (blockDim.x >> 5);
    int tile = blockIdx.x * (blockDim.x >> 5) + waveId;

    for (; tile < numTiles; tile += totalWaves) {
        // ---- load this lane's point with one b96 (two lanes share one point) ----
        const int p16 = tile * 16 + nIdx;
        const int pIdx = p16 < n ? p16 : (n - 1);
        const float3 pt = ((const float3*)x)[pIdx];
        const float q0 = (pt.x - c0 + 1.0f) * 0.5f;
        const float q1 = (pt.y - c1 + 1.0f) * 0.5f;
        const float q2 = (pt.z - c2 + 1.0f) * 0.5f;

        // ---- hash-grid encode straight into the WMMA A-fragment layout ----
        // lanes 0-15: levels {0..7}; lanes 16-31: levels {8..15}
        v16h afrag;
#pragma unroll
        for (int j = 0; j < 8; ++j) {
            const int lA = j;
            const int lB = lA + 8;
            const int res  = loHalf ? kRes[lA] : kRes[lB];
            const unsigned gbase = (unsigned)(loHalf ? lA : lB) << 19;  // level base (elems)
            const bool dA = (lA <= 4), dB = (lB <= 4);   // (res+1)^3 <= 2^19 for levels 0..4
            const bool dense = loHalf ? dA : dB;         // folds per lane-half per j

            const float fres = (float)res;
            const float p0 = q0 * fres, p1 = q1 * fres, p2 = q2 * fres;
            const float f0 = floorf(p0), f1 = floorf(p1), f2 = floorf(p2);
            const float r0 = p0 - f0, r1 = p1 - f1, r2 = p2 - f2;
            const int i0 = (int)f0, i1 = (int)f1, i2 = (int)f2;
            const unsigned s1 = (unsigned)(res + 1);

            // per-axis hoisting: clamped coords, selected index terms, weights
            const unsigned cxv[2] = { (unsigned)i0, (unsigned)min(i0 + 1, res) };
            const unsigned cyv[2] = { (unsigned)i1, (unsigned)min(i1 + 1, res) };
            const unsigned czv[2] = { (unsigned)i2, (unsigned)min(i2 + 1, res) };
            unsigned ty[2], tz[2];
#pragma unroll
            for (int k = 0; k < 2; ++k) {
                ty[k] = dense ? cyv[k] * s1        : cyv[k] * PRIME1;
                tz[k] = dense ? czv[k] * (s1 * s1) : czv[k] * PRIME2;
            }

            // phase 1: all 8 indices; phase 2: all 8 loads in flight; phase 3: FMA
            unsigned idx[8];
#pragma unroll
            for (int c = 0; c < 8; ++c) {
                const int bx = (c >> 2) & 1, by = (c >> 1) & 1, bz = c & 1;
                const unsigned di = cxv[bx] + ty[by] + tz[bz];
                const unsigned hi = (cxv[bx] ^ ty[by] ^ tz[bz]) & TBL_MASK;
                idx[c] = gbase + (dense ? di : hi);      // 23-bit combined offset
            }
            float2 fv[8];
#pragma unroll
            for (int c = 0; c < 8; ++c) fv[c] = tbl2[idx[c]];   // 8 gathers outstanding

            const float wx[2] = { 1.0f - r0, r0 };
            const float wy[2] = { 1.0f - r1, r1 };
            const float wz[2] = { 1.0f - r2, r2 };
            float wxy[4];
#pragma unroll
            for (int k = 0; k < 4; ++k) wxy[k] = wx[k >> 1] * wy[k & 1];

            float a0 = 0.0f, a1 = 0.0f;
#pragma unroll
            for (int c = 0; c < 8; ++c) {
                const float w = wxy[((c >> 2) & 1) * 2 + ((c >> 1) & 1)] * wz[c & 1];
                a0 = fmaf(w, fv[c].x, a0);
                a1 = fmaf(w, fv[c].y, a1);
            }
            afrag[2 * j]     = (_Float16)a0;
            afrag[2 * j + 1] = (_Float16)a1;
        }

        // ---- layer 0: [16x32] x [32x64] via 4 WMMAs (B reloaded from LDS) ----
        v8f h0[4];
#pragma unroll
        for (int t = 0; t < 4; ++t) {
            const v16h bt = *(const v16h*)&wl[t * 16];
            v8f c = {};
            h0[t] = __builtin_amdgcn_wmma_f32_16x16x32_f16(
                false, afrag, false, bt, (short)0, c, false, false);
        }

        // ---- ReLU + stage h0 to LDS col-major [col][row]: one b128 store per tile ----
#pragma unroll
        for (int t = 0; t < 4; ++t) {
            v8h pk;
#pragma unroll
            for (int r = 0; r < 8; ++r) pk[r] = (_Float16)relu_f(h0[t][r]);
            *(v8h*)&st[(t * 16 + nIdx) * 16 + rowOff] = pk;
        }

        __builtin_amdgcn_wave_barrier();   // wave-internal LDS is in-order; block reordering

        // ---- reload h0 as two 16x32 A-fragments (K=0..31, K=32..63) ----
        v16h alo, ahi;
#pragma unroll
        for (int v = 0; v < 8; ++v) {
            const int col = cbase + 2 * v;
            alo[2 * v]     = st[(col)      * 16 + nIdx];
            alo[2 * v + 1] = st[(col + 1)  * 16 + nIdx];
            ahi[2 * v]     = st[(col + 32) * 16 + nIdx];
            ahi[2 * v + 1] = st[(col + 33) * 16 + nIdx];
        }

        // ---- layer 1: [16x64] x [64x64] via 8 WMMAs (K split 32+32) ----
        v8f h1[4];
#pragma unroll
        for (int t = 0; t < 4; ++t) {
            const v16h blo = *(const v16h*)&wl[(4 + t) * 16];
            const v16h bhi = *(const v16h*)&wl[(8 + t) * 16];
            v8f c = {};
            c = __builtin_amdgcn_wmma_f32_16x16x32_f16(
                false, alo, false, blo, (short)0, c, false, false);
            h1[t] = __builtin_amdgcn_wmma_f32_16x16x32_f16(
                false, ahi, false, bhi, (short)0, c, false, false);
        }

        // ---- layer 2: sigma = relu(h1) . W2[:,0], reduce across 16 lanes ----
        float s[8];
#pragma unroll
        for (int r = 0; r < 8; ++r) s[r] = 0.0f;
#pragma unroll
        for (int t = 0; t < 4; ++t)
#pragma unroll
            for (int r = 0; r < 8; ++r)
                s[r] = fmaf(relu_f(h1[t][r]), w2c[t], s[r]);

#pragma unroll
        for (int off = 8; off > 0; off >>= 1)
#pragma unroll
            for (int r = 0; r < 8; ++r)
                s[r] += __shfl_xor(s[r], off, 16);

        if (nIdx == 0) {
            const int base = tile * 16 + rowOff;
            if (base + 8 <= n) {            // fast path: tile fully in range
                float4 o0 = { s[0], s[1], s[2], s[3] };
                float4 o1 = { s[4], s[5], s[6], s[7] };
                *(float4*)&out[base]     = o0;
                *(float4*)&out[base + 4] = o1;
            } else {
#pragma unroll
                for (int r = 0; r < 8; ++r) {
                    const int p = base + r;
                    if (p < n) out[p] = s[r];
                }
            }
        }
    }
}

extern "C" void kernel_launch(void* const* d_in, const int* in_sizes, int n_in,
                              void* d_out, int out_size, void* d_ws, size_t ws_size,
                              hipStream_t stream) {
    const float* x      = (const float*)d_in[0];
    const float* cc     = (const float*)d_in[1];
    const float* tables = (const float*)d_in[2];
    const float* W0     = (const float*)d_in[3];
    const float* W1     = (const float*)d_in[4];
    const float* W2     = (const float*)d_in[5];
    float* out = (float*)d_out;

    const int n = in_sizes[0] / 3;
    const int numTiles = (n + 15) / 16;

    const int wavesPerBlock = 8;               // 256 threads = 8 wave32
    int blocks = (numTiles + wavesPerBlock - 1) / wavesPerBlock;
    if (blocks > 2048) blocks = 2048;          // grid-stride loop covers the rest
    if (blocks < 1) blocks = 1;

    gridnet_sigma_kernel<<<dim3(blocks), dim3(256), 0, stream>>>(
        x, cc, tables, W0, W1, W2, out, n, numTiles);
}
